// Point_Transformer_partseg_80917183857407
// MI455X (gfx1250) — matrix operand order repaired and checked by reading, not compiled, not run on hardware
//
#include <hip/hip_runtime.h>

// ---------------------------------------------------------------------------
// Point-Transformer partseg — CDNA5 (gfx1250) WMMA implementation.
// All GEMMs use v_wmma_f32_16x16x32_f16 (f16 inputs, f32 accumulate), with a
// software-pipelined (double-buffered) K loop + global_prefetch lookahead.
// The fused energy+softmax kernel keeps its 16x4096 strip column-major in LDS
// so each WMMA tile result is one packed ds_store_b128 per lane.
// ---------------------------------------------------------------------------

typedef _Float16 half_t;
typedef __attribute__((ext_vector_type(16))) _Float16 v16h;
typedef __attribute__((ext_vector_type(8)))  _Float16 v8h;
typedef __attribute__((ext_vector_type(8)))  float    v8f;

#define RSQ_BN 0.99999500003749981f   // rsqrt(1 + 1e-5)

// ------------------------------- helpers -----------------------------------

__global__ void cvt_k(const float* __restrict__ src, int sld,
                      half_t* __restrict__ dst, int rows, int cols) {
  int i = blockIdx.x * blockDim.x + threadIdx.x;
  int total = rows * cols;
  if (i >= total) return;
  int r = i / cols, c = i - r * cols;
  dst[i] = (half_t)src[(size_t)r * sld + c];
}

// scale = g*rsqrt(1+eps); shift = (bias? bias*scale : 0) + b
__global__ void mkss_k(const float* __restrict__ g, const float* __restrict__ bb,
                       const float* __restrict__ bias,
                       float* __restrict__ sc, float* __restrict__ sh, int n) {
  int i = blockIdx.x * blockDim.x + threadIdx.x;
  if (i >= n) return;
  float s = g[i] * RSQ_BN;
  sc[i] = s;
  sh[i] = (bias ? bias[i] * s : 0.f) + bb[i];
}

// stem: h1 = relu(bn1(w1 @ x[:,9:12,:]))  -> transposed f16 [B*N][128]
__global__ void stem_k(const float* __restrict__ x, const float* __restrict__ w1,
                       const float* __restrict__ sc, const float* __restrict__ sh,
                       half_t* __restrict__ outT) {
  int gp = blockIdx.x * blockDim.x + threadIdx.x;
  if (gp >= 8192) return;
  int b = gp >> 12, n = gp & 4095;
  const float* xb = x + (size_t)b * 15 * 4096 + n;
  float x0 = xb[9 * 4096], x1 = xb[10 * 4096], x2 = xb[11 * 4096];
  half_t* o = outT + (size_t)gp * 128;
  #pragma unroll 4
  for (int m = 0; m < 128; ++m) {
    float a = w1[3 * m] * x0 + w1[3 * m + 1] * x1 + w1[3 * m + 2] * x2;
    a = a * sc[m] + sh[m];
    o[m] = (half_t)fmaxf(a, 0.f);
  }
}

// --------------------------- generic WMMA GEMM ------------------------------
// out[M][n] = A[M][K] @ Bt[n][K]^T  (Bt is "transposed": K contiguous per point)
// block = 128 threads (4 waves); wave tile = (MT*16) x (NT*16).
// K must be a multiple of 64 (double-buffered pipeline).

enum { EP_PLAIN = 0, EP_BIAS_RM = 1, EP_BN_RELU = 2, EP_BN_LRELU = 3,
       EP_RESID = 4, EP_ATTN = 5 };

template<int MT>
__device__ __forceinline__ void load_af(v16h (&af)[MT], const half_t* Ab,
                                        int ar, int ao, int K, int k0) {
  #pragma unroll
  for (int i = 0; i < MT; ++i) {
    const half_t* p = Ab + (size_t)(ar + i * 16) * K + (k0 + ao);
    v8h lo = *(const v8h*)p;
    v8h hi = *(const v8h*)(p + 8);
    af[i] = __builtin_shufflevector(lo, hi, 0,1,2,3,4,5,6,7,8,9,10,11,12,13,14,15);
  }
}

template<int NT>
__device__ __forceinline__ void load_bf(v16h (&bf)[NT], const half_t* Bb,
                                        int br, int bo, int K, int k0) {
  #pragma unroll
  for (int j = 0; j < NT; ++j)
    bf[j] = *(const v16h*)(Bb + (size_t)(br + j * 16) * K + (k0 + bo));
}

template<int MT, int NT>
__device__ __forceinline__ void mma(v8f (&acc)[MT][NT], const v16h (&af)[MT],
                                    const v16h (&bf)[NT]) {
  #pragma unroll
  for (int i = 0; i < MT; ++i)
    #pragma unroll
    for (int j = 0; j < NT; ++j)
      acc[i][j] = __builtin_amdgcn_wmma_f32_16x16x32_f16(
          false, af[i], false, bf[j], (short)0, acc[i][j], false, false);
}

template<int MT, int NT, int EP>
__global__ void __launch_bounds__(128)
gemm_k(const half_t* __restrict__ A, size_t aBatch, int K,
       const half_t* __restrict__ Bt,
       int M, int Nn,
       half_t* __restrict__ outT, int ldo,
       half_t* __restrict__ outRM,
       const float* __restrict__ bias,
       const float* __restrict__ scale,
       const float* __restrict__ shift, int shiftBatch,
       const float* __restrict__ invcol,
       half_t* __restrict__ resid,
       half_t* __restrict__ feats, int fOff) {
  const int lane = threadIdx.x & 31;
  const int wave = threadIdx.x >> 5;
  const int b = blockIdx.z;
  const int n0 = (blockIdx.x * 4 + wave) * (NT * 16);
  const int m0 = blockIdx.y * (MT * 16);
  if (n0 >= Nn) return;   // wave-uniform exit (EXEC stays all-1s for WMMA)

  const half_t* Ab = A + (size_t)b * aBatch;
  const half_t* Bb = Bt + (size_t)b * (size_t)Nn * (size_t)K;

  // 16-bit WMMA operand k order used here (contiguous, same for A and B):
  //  A: lane = row M (lane&15); halves 0..15 = K base..base+15,
  //     base = lane<16 ? 0 : 16
  //  B: lane = col N (lane&15); halves 0..15 = K 0..15 (lanes<16) / 16..31 (lanes>=16)
  const int ar = m0 + (lane & 15);
  const int ao = (lane & 16) ? 16 : 0;
  const int br = n0 + (lane & 15);
  const int bo = (lane & 16) ? 16 : 0;

  v8f acc[MT][NT] = {};
  v16h af0[MT], bf0[NT], af1[MT], bf1[NT];

  // Software-pipelined K loop: loads for block k+32 issue before the WMMAs of
  // block k, so s_wait only covers the previous prefetch; global_prefetch_b8
  // runs 256 elements (512B) ahead on the streaming B operand.
  load_af<MT>(af0, Ab, ar, ao, K, 0);
  load_bf<NT>(bf0, Bb, br, bo, K, 0);
  for (int k0 = 0; k0 < K; k0 += 64) {
    load_af<MT>(af1, Ab, ar, ao, K, k0 + 32);
    load_bf<NT>(bf1, Bb, br, bo, K, k0 + 32);
    if (k0 + 256 < K) {
      #pragma unroll
      for (int j = 0; j < NT; ++j)
        __builtin_prefetch((const void*)(Bb + (size_t)(br + j * 16) * K +
                                         (k0 + 256 + bo)), 0, 1);
    }
    mma<MT, NT>(acc, af0, bf0);
    if (k0 + 64 < K) {
      load_af<MT>(af0, Ab, ar, ao, K, k0 + 64);
      load_bf<NT>(bf0, Bb, br, bo, K, k0 + 64);
    }
    mma<MT, NT>(acc, af1, bf1);
  }

  // D layout: VGPR r -> row (r + (lane<16?0:8)), col = lane&15
  const int mrB = (lane & 16) ? 8 : 0;
  const int nc = lane & 15;
  #pragma unroll
  for (int j = 0; j < NT; ++j) {
    const int n = n0 + j * 16 + nc;
    const size_t gn = (size_t)b * Nn + n;
    #pragma unroll
    for (int i = 0; i < MT; ++i) {
      const int mb = m0 + i * 16 + mrB;
      if (EP == EP_PLAIN) {
        v8h o;
        #pragma unroll
        for (int r = 0; r < 8; ++r) o[r] = (half_t)acc[i][j][r];
        *(v8h*)(outT + gn * (size_t)ldo + mb) = o;
      } else if (EP == EP_BIAS_RM) {
        #pragma unroll
        for (int r = 0; r < 8; ++r) {
          float v = acc[i][j][r] + bias[mb + r];
          outRM[((size_t)b * M + mb + r) * (size_t)Nn + n] = (half_t)v;
        }
      } else if (EP == EP_BN_RELU || EP == EP_BN_LRELU) {
        const float* sh = shift + (size_t)(shiftBatch ? b * shiftBatch : 0);
        v8h o;
        #pragma unroll
        for (int r = 0; r < 8; ++r) {
          float v = acc[i][j][r] * scale[mb + r] + sh[mb + r];
          v = (EP == EP_BN_RELU) ? fmaxf(v, 0.f) : (v > 0.f ? v : 0.2f * v);
          o[r] = (half_t)v;
        }
        *(v8h*)(outT + gn * (size_t)ldo + mb) = o;
      } else if (EP == EP_RESID) {
        v8h h = *(v8h*)(resid + gn * 128 + mb);
        v8h o;
        #pragma unroll
        for (int r = 0; r < 8; ++r) {
          float v = acc[i][j][r] * scale[mb + r] + shift[mb + r];
          v = fmaxf(v, 0.f) + (float)h[r];
          o[r] = (half_t)v;
        }
        *(v8h*)(resid + gn * 128 + mb) = o;                 // h += relu(bn(...))
        *(v8h*)(feats + gn * 512 + fOff + mb) = o;          // concat feature slab
      } else { // EP_ATTN: d = h - (acc * invcol[n])
        v8h h = *(v8h*)(resid + gn * 128 + mb);
        float ic = invcol[gn];
        v8h o;
        #pragma unroll
        for (int r = 0; r < 8; ++r)
          o[r] = (half_t)((float)h[r] - acc[i][j][r] * ic);
        *(v8h*)(outT + gn * (size_t)ldo + mb) = o;
      }
    }
  }
}

// ----------------- fused energy (q^T q) + row softmax -----------------------
// One block = one 16-row strip of the 4096x4096 attention matrix of batch b.
// The strip lives COLUMN-major in dynamic LDS: e[j][i], j=0..4095, i=0..15
// (128 KB; WGP has 320 KB). Each WMMA tile result is 8 contiguous halves per
// lane -> one packed ds_store_b128; the Pn^T write pass reads one contiguous
// 32B LDS vector per column. The j-tile loop is double-buffered so the B
// fragment load for tile jt+8 issues before the WMMA of tile jt.
__global__ void __launch_bounds__(256)
energy_softmax_k(const half_t* __restrict__ qT, half_t* __restrict__ PnT) {
  extern __shared__ char dynsmem[];
  half_t* e = (half_t*)dynsmem;                 // [4096][16] column-major strip
  __shared__ float red[16][17];
  __shared__ float rowmax[16], rinv[16];

  const int tid = threadIdx.x;
  const int lane = tid & 31;
  const int wave = tid >> 5;
  const int b = blockIdx.y;
  const int i0 = blockIdx.x * 16;
  const half_t* qb = qT + (size_t)b * 4096 * 32;

  // A fragment: rows i0..i0+15, K=32 (one WMMA step)
  const half_t* pa = qb + (size_t)(i0 + (lane & 15)) * 32 + ((lane & 16) ? 16 : 0);
  v8h alo = *(const v8h*)pa;
  v8h ahi = *(const v8h*)(pa + 8);
  v16h af = __builtin_shufflevector(alo, ahi, 0,1,2,3,4,5,6,7,8,9,10,11,12,13,14,15);

  const int rb = (lane & 16) ? 8 : 0;
  const int bcol = lane & 15;
  const int boff = (lane & 16) ? 16 : 0;

  v16h bf = *(const v16h*)(qb + (size_t)(wave * 16 + bcol) * 32 + boff);
  for (int jt = wave; jt < 256; jt += 8) {
    v16h bfn;
    if (jt + 8 < 256)
      bfn = *(const v16h*)(qb + (size_t)((jt + 8) * 16 + bcol) * 32 + boff);
    v8f c = {};
    c = __builtin_amdgcn_wmma_f32_16x16x32_f16(false, af, false, bf, (short)0, c,
                                               false, false);
    // column (jt*16 + bcol), rows rb..rb+7: contiguous -> one b128 DS store
    v8h o;
    #pragma unroll
    for (int r = 0; r < 8; ++r) o[r] = (half_t)c[r];
    *(v8h*)(e + (size_t)(jt * 16 + bcol) * 16 + rb) = o;
    bf = bfn;
  }
  __syncthreads();

  // row max (16 rows x 16 chunks of 256 cols); strip is e[j*16 + i]
  const int row = tid & 15, chunk = tid >> 4;
  float m = -3.4e38f;
  for (int jj = chunk * 256; jj < chunk * 256 + 256; ++jj)
    m = fmaxf(m, (float)e[(size_t)jj * 16 + row]);
  red[row][chunk] = m;
  __syncthreads();
  if (tid < 16) {
    float mm = -3.4e38f;
    #pragma unroll
    for (int c = 0; c < 16; ++c) mm = fmaxf(mm, red[tid][c]);
    rowmax[tid] = mm;
  }
  __syncthreads();

  // exp + row sum; overwrite strip with unnormalized P (<=1, f16-safe)
  float rm = rowmax[row], s = 0.f;
  for (int jj = chunk * 256; jj < chunk * 256 + 256; ++jj) {
    float p = __expf((float)e[(size_t)jj * 16 + row] - rm);
    e[(size_t)jj * 16 + row] = (half_t)p;
    s += p;
  }
  red[row][chunk] = s;
  __syncthreads();
  if (tid < 16) {
    float ss = 0.f;
    #pragma unroll
    for (int c = 0; c < 16; ++c) ss += red[tid][c];
    rinv[tid] = 1.f / ss;
  }
  __syncthreads();

  // write Pn^T: one contiguous 32B LDS read + 32B global write per column
  for (int j = tid; j < 4096; j += 256) {
    v16h pcol = *(const v16h*)(e + (size_t)j * 16);
    v16h o;
    #pragma unroll
    for (int i = 0; i < 16; ++i)
      o[i] = (half_t)((float)pcol[i] * rinv[i]);
    *(v16h*)(PnT + ((size_t)b * 4096 + j) * 4096 + i0) = o;
  }
}

// invcol[b*N + j] = 1 / (1e-9 + sum_i Pn^T[j][i]); one wave per j (coalesced)
__global__ void colsum_k(const half_t* __restrict__ PnT, float* __restrict__ invcol,
                         int total) {
  int gw = (blockIdx.x * blockDim.x + threadIdx.x) >> 5;
  int lane = threadIdx.x & 31;
  if (gw >= total) return;
  const half_t* p = PnT + (size_t)gw * 4096;
  float s = 0.f;
  for (int i0 = lane * 8; i0 < 4096; i0 += 256) {
    v8h x = *(const v8h*)(p + i0);
    #pragma unroll
    for (int r = 0; r < 8; ++r) s += (float)x[r];
  }
  #pragma unroll
  for (int off = 16; off > 0; off >>= 1) s += __shfl_down(s, off, 32);
  if (lane == 0) invcol[gw] = 1.f / (1e-9f + s);
}

// global max/mean over points; threads = channels (coalesced column reads)
__global__ void pool_k(const half_t* __restrict__ hfT, float* __restrict__ pmax,
                       float* __restrict__ pavg) {
  int c = blockIdx.x * blockDim.x + threadIdx.x;
  int b = blockIdx.y;
  if (c >= 1024) return;
  const half_t* p = hfT + (size_t)b * 4096 * 1024 + c;
  float mx = -3.4e38f, sm = 0.f;
  for (int n = 0; n < 4096; ++n) {
    float v = (float)p[(size_t)n * 1024];
    mx = fmaxf(mx, v);
    sm += v;
  }
  pmax[b * 1024 + c] = mx;
  pavg[b * 1024 + c] = sm * (1.f / 4096.f);
}

// per-batch effective shift for ws1: broadcast max/avg channels fold into bias
__global__ void shift1_k(const float* __restrict__ ws1, const float* __restrict__ bs1,
                         const float* __restrict__ g, const float* __restrict__ bb,
                         const float* __restrict__ pmax, const float* __restrict__ pavg,
                         float* __restrict__ shift1) {
  int m1 = blockIdx.x * blockDim.x + threadIdx.x;
  int b = blockIdx.y;
  if (m1 >= 512) return;
  const float* wrow = ws1 + (size_t)m1 * 3072;
  float a = bs1[m1];
  for (int c = 0; c < 1024; ++c)
    a += wrow[1024 + c] * pmax[b * 1024 + c] + wrow[2048 + c] * pavg[b * 1024 + c];
  float s = g[m1] * RSQ_BN;
  shift1[b * 512 + m1] = a * s + bb[m1];
}

// ws3 (8x256) + log_softmax over 8 classes; out[b][n][k] f32
__global__ void head_k(const half_t* __restrict__ s2T, const float* __restrict__ w3,
                       const float* __restrict__ b3, float* __restrict__ out) {
  int gp = blockIdx.x * blockDim.x + threadIdx.x;
  if (gp >= 8192) return;
  const half_t* s = s2T + (size_t)gp * 256;
  float l[8];
  #pragma unroll
  for (int k = 0; k < 8; ++k) {
    float a = b3[k];
    for (int c = 0; c < 256; ++c) a += w3[k * 256 + c] * (float)s[c];
    l[k] = a;
  }
  float m = l[0];
  #pragma unroll
  for (int k = 1; k < 8; ++k) m = fmaxf(m, l[k]);
  float se = 0.f;
  #pragma unroll
  for (int k = 0; k < 8; ++k) se += __expf(l[k] - m);
  float lse = m + __logf(se);
  #pragma unroll
  for (int k = 0; k < 8; ++k) out[(size_t)gp * 8 + k] = l[k] - lse;
}

// ------------------------------- launcher -----------------------------------

extern "C" void kernel_launch(void* const* d_in, const int* in_sizes, int n_in,
                              void* d_out, int out_size, void* d_ws, size_t ws_size,
                              hipStream_t stream) {
  (void)in_sizes; (void)n_in; (void)out_size; (void)ws_size;
  constexpr int Npt = 4096, B = 2;
  constexpr size_t BN = (size_t)B * Npt;

  const float* x      = (const float*)d_in[0];
  const float* w1     = (const float*)d_in[1];
  const float* bn1_g  = (const float*)d_in[2];
  const float* bn1_b  = (const float*)d_in[3];
  const float* w2     = (const float*)d_in[4];
  const float* bn2_g  = (const float*)d_in[5];
  const float* bn2_b  = (const float*)d_in[6];
  const float* sa_wqk = (const float*)d_in[7];
  const float* sa_wv  = (const float*)d_in[8];
  const float* sa_bv  = (const float*)d_in[9];
  const float* sa_wt  = (const float*)d_in[10];
  const float* sa_bt  = (const float*)d_in[11];
  const float* sa_bng = (const float*)d_in[12];
  const float* sa_bnb = (const float*)d_in[13];
  const float* wf     = (const float*)d_in[14];
  const float* bnf_g  = (const float*)d_in[15];
  const float* bnf_b  = (const float*)d_in[16];
  const float* ws1    = (const float*)d_in[17];
  const float* bs1    = (const float*)d_in[18];
  const float* bns1_g = (const float*)d_in[19];
  const float* bns1_b = (const float*)d_in[20];
  const float* ws2    = (const float*)d_in[21];
  const float* bs2    = (const float*)d_in[22];
  const float* bns2_g = (const float*)d_in[23];
  const float* bns2_b = (const float*)d_in[24];
  const float* ws3    = (const float*)d_in[25];
  const float* bs3    = (const float*)d_in[26];
  float* out = (float*)d_out;

  // ---- workspace carve-up ----
  char* W = (char*)d_ws;
  size_t off = 0;
  auto take = [&](size_t n) { char* p = W + off; off = (off + n + 255) & ~(size_t)255; return p; };
  half_t* HT   = (half_t*)take(BN * 128 * 2);              // h (transposed f16)
  half_t* QT   = (half_t*)take(BN * 32 * 2);               // q^T
  half_t* VRM  = (half_t*)take((size_t)B * 128 * Npt * 2); // v row-major
  half_t* DT   = (half_t*)take(BN * 128 * 2);              // d = h - x_r (and stem tmp)
  half_t* FE   = (half_t*)take(BN * 512 * 2);              // concat feats
  half_t* HF   = (half_t*)take(BN * 1024 * 2);             // after wf+leaky
  half_t* S1   = (half_t*)take(BN * 512 * 2);
  half_t* S2   = (half_t*)take(BN * 256 * 2);
  half_t* PNT  = (half_t*)take((size_t)B * Npt * Npt * 2); // Pn^T (64 MB)
  float*  INVC = (float*)take(BN * 4);
  float*  PMAX = (float*)take((size_t)B * 1024 * 4);
  float*  PAVG = (float*)take((size_t)B * 1024 * 4);
  float*  SHF1 = (float*)take((size_t)B * 512 * 4);
  half_t* W2H  = (half_t*)take(128 * 128 * 2);
  half_t* WQKH = (half_t*)take(4 * 32 * 128 * 2);
  half_t* WVH  = (half_t*)take(4 * 128 * 128 * 2);
  half_t* WTH  = (half_t*)take(4 * 128 * 128 * 2);
  half_t* WFH  = (half_t*)take((size_t)1024 * 512 * 2);
  half_t* WS1H = (half_t*)take((size_t)512 * 1024 * 2);
  half_t* WS2H = (half_t*)take((size_t)256 * 512 * 2);
  float* SC1  = (float*)take(128 * 4);  float* SH1  = (float*)take(128 * 4);
  float* SC2  = (float*)take(128 * 4);  float* SH2  = (float*)take(128 * 4);
  float* SASC = (float*)take(512 * 4);  float* SASH = (float*)take(512 * 4);
  float* SCF  = (float*)take(1024 * 4); float* SHFF = (float*)take(1024 * 4);
  float* SCS1 = (float*)take(512 * 4);  float* SHD1 = (float*)take(512 * 4);
  float* SCS2 = (float*)take(256 * 4);  float* SHS2 = (float*)take(256 * 4);

  auto cvt = [&](const float* s, int sld, half_t* d, int r, int c) {
    int tot = r * c;
    cvt_k<<<dim3((tot + 255) / 256), dim3(256), 0, stream>>>(s, sld, d, r, c);
  };

  // ---- weight conversion + bn folding ----
  cvt(w2, 128, W2H, 128, 128);
  cvt(sa_wqk, 128, WQKH, 4 * 32, 128);
  cvt(sa_wv, 128, WVH, 4 * 128, 128);
  cvt(sa_wt, 128, WTH, 4 * 128, 128);
  cvt(wf, 512, WFH, 1024, 512);
  cvt(ws1, 3072, WS1H, 512, 1024);   // only the non-broadcast K block
  cvt(ws2, 512, WS2H, 256, 512);
  mkss_k<<<dim3(1), dim3(128), 0, stream>>>(bn1_g, bn1_b, nullptr, SC1, SH1, 128);
  mkss_k<<<dim3(1), dim3(128), 0, stream>>>(bn2_g, bn2_b, nullptr, SC2, SH2, 128);
  mkss_k<<<dim3(2), dim3(256), 0, stream>>>(sa_bng, sa_bnb, sa_bt, SASC, SASH, 512);
  mkss_k<<<dim3(4), dim3(256), 0, stream>>>(bnf_g, bnf_b, nullptr, SCF, SHFF, 1024);
  mkss_k<<<dim3(2), dim3(256), 0, stream>>>(bns1_g, bns1_b, nullptr, SCS1, SHD1, 512);
  mkss_k<<<dim3(1), dim3(256), 0, stream>>>(bns2_g, bns2_b, bs2, SCS2, SHS2, 256);

  // ---- stem ----
  stem_k<<<dim3(32), dim3(256), 0, stream>>>(x, w1, SC1, SH1, DT);
  // h = relu(bn2(w2 @ h1))
  gemm_k<2, 4, EP_BN_RELU><<<dim3(16, 4, B), dim3(128), 0, stream>>>(
      W2H, 0, 128, DT, 128, Npt, HT, 128, nullptr, nullptr, SC2, SH2, 0,
      nullptr, nullptr, nullptr, 0);

  // ---- 4 self-attention layers ----
  for (int L = 0; L < 4; ++L) {
    // q^T = (wqk @ h)^T   [BN][32]
    gemm_k<2, 4, EP_PLAIN><<<dim3(16, 1, B), dim3(128), 0, stream>>>(
        WQKH + (size_t)L * 32 * 128, 0, 128, HT, 32, Npt, QT, 32,
        nullptr, nullptr, nullptr, nullptr, 0, nullptr, nullptr, nullptr, 0);
    // v = wv @ h + bv   row-major [b][128][N]
    gemm_k<2, 4, EP_BIAS_RM><<<dim3(16, 4, B), dim3(128), 0, stream>>>(
        WVH + (size_t)L * 128 * 128, 0, 128, HT, 128, Npt, nullptr, 0, VRM,
        sa_bv + L * 128, nullptr, nullptr, 0, nullptr, nullptr, nullptr, 0);
    // energy strip + softmax -> Pn^T  (128 KB dynamic LDS per block)
    energy_softmax_k<<<dim3(256, B), dim3(256), 16 * 4096 * 2, stream>>>(QT, PNT);
    // column sums -> invcol
    colsum_k<<<dim3(1024), dim3(256), 0, stream>>>(PNT, INVC, (int)BN);
    // d = h - (v @ Pn) * invcol   (the big K=4096 GEMM)
    gemm_k<2, 4, EP_ATTN><<<dim3(16, 4, B), dim3(128), 0, stream>>>(
        VRM, (size_t)128 * Npt, Npt, PNT, 128, Npt, DT, 128, nullptr, nullptr,
        nullptr, nullptr, 0, INVC, HT, nullptr, 0);
    // h += relu(bn(wt @ d + bt)); also write feats slab
    gemm_k<2, 4, EP_RESID><<<dim3(16, 4, B), dim3(128), 0, stream>>>(
        WTH + (size_t)L * 128 * 128, 0, 128, DT, 128, Npt, nullptr, 0, nullptr,
        nullptr, SASC + L * 128, SASH + L * 128, 0, nullptr, HT, FE, 128 * L);
  }

  // ---- head ----
  // hf = leaky_relu(bnf(wf @ feats), 0.2)
  gemm_k<2, 4, EP_BN_LRELU><<<dim3(16, 32, B), dim3(128), 0, stream>>>(
      WFH, 0, 512, FE, 1024, Npt, HF, 1024, nullptr, nullptr, SCF, SHFF, 0,
      nullptr, nullptr, nullptr, 0);
  pool_k<<<dim3(4, B), dim3(256), 0, stream>>>(HF, PMAX, PAVG);
  shift1_k<<<dim3(2, B), dim3(256), 0, stream>>>(ws1, bs1, bns1_g, bns1_b,
                                                 PMAX, PAVG, SHF1);
  // s1 = relu(bn(ws1[:, :1024] @ hf + per-batch folded bias))
  gemm_k<2, 4, EP_BN_RELU><<<dim3(16, 16, B), dim3(128), 0, stream>>>(
      WS1H, 0, 1024, HF, 512, Npt, S1, 512, nullptr, nullptr, SCS1, SHF1, 512,
      nullptr, nullptr, nullptr, 0);
  // s2 = relu(bn(ws2 @ s1 + bs2))
  gemm_k<2, 4, EP_BN_RELU><<<dim3(16, 8, B), dim3(128), 0, stream>>>(
      WS2H, 0, 512, S1, 256, Npt, S2, 256, nullptr, nullptr, SCS2, SHS2, 0,
      nullptr, nullptr, nullptr, 0);
  // logits + log_softmax -> out (B, N, 8)
  head_k<<<dim3(32), dim3(256), 0, stream>>>(S2, ws3, bs3, out);
}
